// PoseSSM_module_73813307949400
// MI455X (gfx1250) — matrix operand order repaired and checked by reading, not compiled, not run on hardware
//
#include <hip/hip_runtime.h>
#include <hip/hip_bf16.h>
#include <stdint.h>

// ---------------------------------------------------------------------------
// PoseSSM on MI455X (gfx1250): bf16 WMMA GEMMs + fp32 scalar scan.
// D_MODEL=512, L=17, D_INNER=1024, D_STATE=4, DT_RANK=32, B=4096 -> BL=69632.
// Round 3: async-to-LDS builtin confirmed present; fix its pointer element
// type (b128 wants int-vector(16B) pointers in AS1/AS3).
// ---------------------------------------------------------------------------

typedef unsigned short u16t;
typedef __bf16 v16bf __attribute__((ext_vector_type(16)));
typedef float  v8f   __attribute__((ext_vector_type(8)));

// ---- gfx1250 async-to-LDS path (opportunistic, compile-safe) --------------
#if defined(__has_builtin)
#if __has_builtin(__builtin_amdgcn_global_load_async_to_lds_b128)
#define USE_ASYNC_LDS 1
#endif
#endif
#ifndef USE_ASYNC_LDS
#define USE_ASYNC_LDS 0
#endif

#if USE_ASYNC_LDS
typedef int vsi4 __attribute__((__vector_size__(16)));
typedef vsi4 vsi4_as1 __attribute__((address_space(1)));
typedef vsi4 vsi4_as3 __attribute__((address_space(3)));
#endif

__device__ __forceinline__ void copy16_to_lds(u16t* l, const u16t* g) {
#if USE_ASYNC_LDS
    __builtin_amdgcn_global_load_async_to_lds_b128(
        (vsi4_as1*)(u16t*)g, (vsi4_as3*)l, 0, 0);
#else
    *(uint4*)l = *(const uint4*)g;
#endif
}

__device__ __forceinline__ void wait_async_lds() {
#if USE_ASYNC_LDS
#if __has_builtin(__builtin_amdgcn_s_wait_asynccnt)
    __builtin_amdgcn_s_wait_asynccnt(0);
#else
    asm volatile("s_wait_asynccnt 0x0" ::: "memory");
#endif
#endif
}

// GRAPH^-1 == HOP (verified: GRAPH[l]=l' <=> HOP[l']=l)
__constant__ int c_HOP[17]   = {0,1,4,7,2,5,8,3,6,9,11,14,10,12,15,13,16};
__constant__ int c_BPIDX[17] = {0,1,2, 0,1,2, 0,1,2, 0, 3,4, 0, 3,4, 3,4};

// ---- bf16 helpers (RNE), storage as u16 so no scalar __bf16 arithmetic ----
__device__ __forceinline__ float bf2f(u16t u) {
    union { unsigned int i; float f; } x; x.i = ((unsigned int)u) << 16; return x.f;
}
__device__ __forceinline__ u16t f2bf(float f) {
    union { float f; unsigned int i; } x; x.f = f;
    unsigned int u = x.i;
    return (u16t)((u + 0x7FFFu + ((u >> 16) & 1u)) >> 16);
}

struct U32x8 { uint4 lo, hi; };
__device__ __forceinline__ v16bf make_frag(uint4 lo, uint4 hi) {
    U32x8 t{lo, hi};
    return __builtin_bit_cast(v16bf, t);
}

__device__ __forceinline__ float fast_sigmoid(float t) { return 1.0f / (1.0f + __expf(-t)); }

// ---------------------------------------------------------------------------
// Weight conversion kernels
// ---------------------------------------------------------------------------
__global__ void cvt_bf16_kernel(const float* __restrict__ s, u16t* __restrict__ d, int n) {
    int i = blockIdx.x * 256 + threadIdx.x;
    if (i < n) d[i] = f2bf(s[i]);
}

// x_proj_w (40x1024) -> padded (48x1024) bf16
__global__ void cvt_xproj_pad_kernel(const float* __restrict__ s, u16t* __restrict__ d) {
    int i = blockIdx.x * 256 + threadIdx.x;
    if (i < 48 * 1024) {
        int r = i >> 10, c = i & 1023;
        d[i] = (r < 40) ? f2bf(s[r * 1024 + c]) : (u16t)0;
    }
}

// ---------------------------------------------------------------------------
// Gather(HOP) + bp-embed + LayerNorm -> bf16  (one block per output row)
// ---------------------------------------------------------------------------
__global__ __launch_bounds__(256) void gather_ln_kernel(
    const float* __restrict__ x, const float* __restrict__ bpe,
    const float* __restrict__ lnw, const float* __restrict__ lnb,
    u16t* __restrict__ Xln)
{
    const int m  = blockIdx.x;
    const int b  = m / 17;
    const int lp = m - b * 17;
    const float* xr = x   + ((size_t)b * 17 + c_HOP[lp]) * 512;
    const float* br = bpe + (size_t)c_BPIDX[lp] * 512;
    const int t = threadIdx.x;

    float v0 = xr[t]       + br[t];
    float v1 = xr[t + 256] + br[t + 256];

    __shared__ float red[256];
    red[t] = v0 + v1;
    __syncthreads();
    for (int s = 128; s > 0; s >>= 1) { if (t < s) red[t] += red[t + s]; __syncthreads(); }
    const float mu = red[0] * (1.0f / 512.0f);
    __syncthreads();

    float d0 = v0 - mu, d1 = v1 - mu;
    red[t] = d0 * d0 + d1 * d1;
    __syncthreads();
    for (int s = 128; s > 0; s >>= 1) { if (t < s) red[t] += red[t + s]; __syncthreads(); }
    const float rs = rsqrtf(red[0] * (1.0f / 512.0f) + 1e-5f);

    u16t* out = Xln + (size_t)m * 512;
    out[t]       = f2bf(d0 * rs * lnw[t]       + lnb[t]);
    out[t + 256] = f2bf(d1 * rs * lnw[t + 256] + lnb[t + 256]);
}

// ---------------------------------------------------------------------------
// Generic 128x128 block-tile WMMA GEMM: C[m,n] = sum_k A[m,k] * W[n,k]
// A: (M x K) bf16 row-major, W: (N x K) bf16 row-major (K-contiguous both).
// 256 threads = 8 waves; wave tile = 64(M) x 32(N) = 4x2 WMMA accumulators.
// K staged through double-buffered LDS in 32-wide steps; async-to-LDS copies
// when available; exactly one barrier per K step.
// ---------------------------------------------------------------------------
template <typename Ep>
__global__ __launch_bounds__(256) void wmma_gemm128(
    const u16t* __restrict__ A, const u16t* __restrict__ W,
    int M, int N, int K, Ep ep)
{
    __shared__ __align__(16) u16t sA[2][128 * 32];
    __shared__ __align__(16) u16t sB[2][128 * 32];

    const int tid  = threadIdx.x;
    const int lane = tid & 31;
    const int wave = tid >> 5;
    const int wm   = wave >> 2;   // 0..1  (64-row half)
    const int wn   = wave & 3;    // 0..3  (32-col quarter)
    const int m0   = blockIdx.y * 128;
    const int n0   = blockIdx.x * 128;

    const int kh = lane >> 4;     // K half selector per ISA layout
    const int ll = lane & 15;

    v8f acc[4][2];
    const v8f vzero = {0.f,0.f,0.f,0.f,0.f,0.f,0.f,0.f};
#pragma unroll
    for (int i = 0; i < 4; ++i)
#pragma unroll
        for (int j = 0; j < 2; ++j) acc[i][j] = vzero;

    // 16-B chunk fill: chunk c covers lds ushort [c*8, c*8+8)
    auto fill = [&](int buf, int k0) {
#pragma unroll
        for (int i = 0; i < 2; ++i) {
            const int c   = tid + i * 256;   // 0..511
            const int row = c >> 2;
            const int seg = c & 3;
            copy16_to_lds(&sA[buf][c * 8], A + (size_t)(m0 + row) * K + k0 + seg * 8);
            copy16_to_lds(&sB[buf][c * 8], W + (size_t)(n0 + row) * K + k0 + seg * 8);
        }
    };

    const int nk = K >> 5;
    fill(0, 0);

    for (int it = 0; it < nk; ++it) {
        wait_async_lds();      // my async copies for buf 'it&1' complete
        __syncthreads();       // everyone's copies visible; prev reads done
        if (it + 1 < nk) fill((it + 1) & 1, (it + 1) * 32);
        const u16t* cA = sA[it & 1];
        const u16t* cB = sB[it & 1];

        // ---- B fragments: col=lane, K {kh*8..}+{16+kh*8..}, 2 per VGPR ----
        v16bf bfrag[2];
#pragma unroll
        for (int nt = 0; nt < 2; ++nt) {
            const u16t* p = cB + (wn * 32 + nt * 16 + ll) * 32 + kh * 8;
            bfrag[nt] = make_frag(*(const uint4*)p, *(const uint4*)(p + 16));
        }
        // ---- A fragments: row=lane, K {kh*8..}+{16+kh*8..} ----
#pragma unroll
        for (int mt = 0; mt < 4; ++mt) {
            const u16t* p = cA + (wm * 64 + mt * 16 + ll) * 32 + kh * 8;
            v16bf afrag = make_frag(*(const uint4*)p, *(const uint4*)(p + 16));
#pragma unroll
            for (int nt = 0; nt < 2; ++nt)
                acc[mt][nt] = __builtin_amdgcn_wmma_f32_16x16x32_bf16(
                    false, afrag, false, bfrag[nt], (short)0, acc[mt][nt], false, false);
        }
    }

    // ---- epilogue: C layout VGPR j -> M=j (lanes 0-15) / M=j+8 (16-31) ----
#pragma unroll
    for (int mt = 0; mt < 4; ++mt)
#pragma unroll
        for (int nt = 0; nt < 2; ++nt)
#pragma unroll
            for (int j = 0; j < 8; ++j) {
                const int m = m0 + wm * 64 + mt * 16 + kh * 8 + j;
                const int n = n0 + wn * 32 + nt * 16 + ll;
                ep(m, n, acc[mt][nt][j]);
            }
}

// ---- Epilogue functors ----------------------------------------------------
struct EpInProj {   // xz split: silu(xm*conv_w+conv_b) -> bf16; silu(z) -> bf16
    const float* cw; const float* cb; u16t* xm; u16t* sz;
    __device__ void operator()(int m, int n, float v) const {
        if (n < 1024) {
            float t = v * cw[n] + cb[n];
            xm[(size_t)m * 1024 + n] = f2bf(t * fast_sigmoid(t));
        } else {
            sz[(size_t)m * 1024 + (n - 1024)] = f2bf(v * fast_sigmoid(v));
        }
    }
};
struct EpDt {       // softplus(dt_pre + dt_proj_b) -> bf16
    const float* bias; u16t* dt;
    __device__ void operator()(int m, int n, float v) const {
        float t = v + bias[n];
        float sp = (t > 20.0f) ? t : log1pf(__expf(t));
        dt[(size_t)m * 1024 + n] = f2bf(sp);
    }
};
struct EpOut {      // out[b, HOP[l'], d] = x[...] + mamba_out[b, l', d]
    const float* x; float* out;
    __device__ void operator()(int m, int n, float v) const {
        const int b  = m / 17;
        const int lp = m - b * 17;
        const size_t row = (size_t)b * 17 + c_HOP[lp];  // GRAPH^-1 == HOP
        out[row * 512 + n] = x[row * 512 + n] + v;
    }
};

// ---------------------------------------------------------------------------
// x_proj GEMM: x_dbl = xm(BLx1024) * W2p^T (48x1024), small-N direct-frag WMMA.
// 8 waves/block, each wave: one 16-row M tile x 3 N tiles (N=48), K loop 1024.
// Epilogue: cols 0..31 -> dt_r bf16; 32..39 -> [B|C] fp32; 40..47 discarded.
// ---------------------------------------------------------------------------
__global__ __launch_bounds__(256) void wmma_gemm_xproj(
    const u16t* __restrict__ A, const u16t* __restrict__ W,
    u16t* __restrict__ dtr, float* __restrict__ BC)
{
    const int tid  = threadIdx.x;
    const int lane = tid & 31;
    const int wave = tid >> 5;
    const int m0   = blockIdx.x * 128 + wave * 16;
    const int kh   = lane >> 4;
    const int ll   = lane & 15;

    const v8f vzero = {0.f,0.f,0.f,0.f,0.f,0.f,0.f,0.f};
    v8f acc[3] = {vzero, vzero, vzero};

    for (int k0 = 0; k0 < 1024; k0 += 32) {
        if (k0 + 32 < 1024)
            __builtin_prefetch(A + (size_t)(m0 + ll) * 1024 + k0 + 32 + kh * 16, 0, 0);
        const u16t* pa = A + (size_t)(m0 + ll) * 1024 + k0 + kh * 8;
        v16bf afrag = make_frag(*(const uint4*)pa, *(const uint4*)(pa + 16));
#pragma unroll
        for (int nt = 0; nt < 3; ++nt) {
            const u16t* pb = W + (size_t)(nt * 16 + ll) * 1024 + k0 + kh * 8;
            v16bf bfrag = make_frag(*(const uint4*)pb, *(const uint4*)(pb + 16));
            acc[nt] = __builtin_amdgcn_wmma_f32_16x16x32_bf16(
                false, afrag, false, bfrag, (short)0, acc[nt], false, false);
        }
    }
#pragma unroll
    for (int nt = 0; nt < 3; ++nt)
#pragma unroll
        for (int j = 0; j < 8; ++j) {
            const int m = m0 + kh * 8 + j;
            const int n = nt * 16 + ll;
            const float v = acc[nt][j];
            if (n < 32)      dtr[(size_t)m * 32 + n]      = f2bf(v);
            else if (n < 40) BC[(size_t)m * 8 + (n - 32)] = v;
        }
}

// ---------------------------------------------------------------------------
// Selective scan: one thread per (b, channel). L=17, D_STATE=4.
// y = (scan_y + xm*D) * silu(z)   (silu(z) precomputed in sz_bf)
// ---------------------------------------------------------------------------
__global__ __launch_bounds__(256) void scan_kernel(
    const u16t* __restrict__ dt_bf, const u16t* __restrict__ xm_bf,
    const float* __restrict__ BC,   const u16t* __restrict__ sz_bf,
    const float* __restrict__ Alog, const float* __restrict__ Dp,
    u16t* __restrict__ y_bf)
{
    const int e = blockIdx.x * 256 + threadIdx.x;   // 0..1023
    const int b = blockIdx.y;

    float As0 = -__expf(Alog[e * 4 + 0]);
    float As1 = -__expf(Alog[e * 4 + 1]);
    float As2 = -__expf(Alog[e * 4 + 2]);
    float As3 = -__expf(Alog[e * 4 + 3]);
    const float Dv = Dp[e];

    float h0 = 0.f, h1 = 0.f, h2 = 0.f, h3 = 0.f;
#pragma unroll
    for (int l = 0; l < 17; ++l) {
        const size_t row = (size_t)b * 17 + l;
        const size_t idx = row * 1024 + e;
        const float dtv = bf2f(dt_bf[idx]);
        const float xv  = bf2f(xm_bf[idx]);
        const float4 Bv = *(const float4*)(BC + row * 8);
        const float4 Cv = *(const float4*)(BC + row * 8 + 4);
        const float dx = dtv * xv;
        h0 = __expf(dtv * As0) * h0 + dx * Bv.x;
        h1 = __expf(dtv * As1) * h1 + dx * Bv.y;
        h2 = __expf(dtv * As2) * h2 + dx * Bv.z;
        h3 = __expf(dtv * As3) * h3 + dx * Bv.w;
        float y = h0 * Cv.x + h1 * Cv.y + h2 * Cv.z + h3 * Cv.w;
        y = (y + xv * Dv) * bf2f(sz_bf[idx]);
        y_bf[idx] = f2bf(y);
    }
}

// ---------------------------------------------------------------------------
// Host launcher
// ---------------------------------------------------------------------------
extern "C" void kernel_launch(void* const* d_in, const int* in_sizes, int n_in,
                              void* d_out, int out_size, void* d_ws, size_t ws_size,
                              hipStream_t stream) {
    const float* x         = (const float*)d_in[0];
    const float* bp_embed  = (const float*)d_in[1];
    const float* ln_w      = (const float*)d_in[2];
    const float* ln_b      = (const float*)d_in[3];
    const float* in_proj_w = (const float*)d_in[4];
    const float* conv_w    = (const float*)d_in[5];
    const float* conv_b    = (const float*)d_in[6];
    const float* x_proj_w  = (const float*)d_in[7];
    const float* dt_proj_w = (const float*)d_in[8];
    const float* dt_proj_b = (const float*)d_in[9];
    const float* A_log     = (const float*)d_in[10];
    const float* Dvec      = (const float*)d_in[11];
    const float* out_proj_w= (const float*)d_in[12];
    float* out = (float*)d_out;

    const int BL = in_sizes[0] / 512;   // B * 17 (= 69632, multiple of 128)
    const int Bn = BL / 17;

    // --- carve workspace (256B-aligned slabs) ---
    uintptr_t p = (uintptr_t)d_ws;
    auto alloc = [&](size_t bytes) -> void* {
        p = (p + 255) & ~(uintptr_t)255;
        void* r = (void*)p;
        p += bytes;
        return r;
    };
    u16t* Xln  = (u16t*)alloc((size_t)BL * 512 * 2);
    u16t* W1   = (u16t*)alloc((size_t)2048 * 512 * 2);
    u16t* W2   = (u16t*)alloc((size_t)48 * 1024 * 2);
    u16t* Wdt  = (u16t*)alloc((size_t)1024 * 32 * 2);
    u16t* Wout = (u16t*)alloc((size_t)512 * 1024 * 2);
    u16t* xm   = (u16t*)alloc((size_t)BL * 1024 * 2);
    u16t* sz   = (u16t*)alloc((size_t)BL * 1024 * 2);
    u16t* dtr  = (u16t*)alloc((size_t)BL * 32 * 2);
    float* BC  = (float*)alloc((size_t)BL * 8 * 4);
    u16t* dtb  = (u16t*)alloc((size_t)BL * 1024 * 2);
    u16t* yb   = (u16t*)alloc((size_t)BL * 1024 * 2);

    // --- 1) weight casts ---
    {
        int n1 = 2048 * 512, n3 = 1024 * 32, n4 = 512 * 1024;
        cvt_bf16_kernel<<<(n1 + 255) / 256, 256, 0, stream>>>(in_proj_w, W1, n1);
        cvt_xproj_pad_kernel<<<(48 * 1024 + 255) / 256, 256, 0, stream>>>(x_proj_w, W2);
        cvt_bf16_kernel<<<(n3 + 255) / 256, 256, 0, stream>>>(dt_proj_w, Wdt, n3);
        cvt_bf16_kernel<<<(n4 + 255) / 256, 256, 0, stream>>>(out_proj_w, Wout, n4);
    }

    // --- 2) gather + bpe + layernorm -> bf16 ---
    gather_ln_kernel<<<BL, 256, 0, stream>>>(x, bp_embed, ln_w, ln_b, Xln);

    // --- 3) in_proj GEMM (M=BL, N=2048, K=512) + silu/conv epilogue ---
    wmma_gemm128<<<dim3(2048 / 128, BL / 128), 256, 0, stream>>>(
        Xln, W1, BL, 2048, 512, EpInProj{conv_w, conv_b, xm, sz});

    // --- 4) x_proj GEMM (N=48, K=1024) -> dt_r bf16, B/C fp32 ---
    wmma_gemm_xproj<<<BL / 128, 256, 0, stream>>>(xm, W2, dtr, BC);

    // --- 5) dt_proj GEMM (M=BL, N=1024, K=32) + softplus ---
    wmma_gemm128<<<dim3(1024 / 128, BL / 128), 256, 0, stream>>>(
        dtr, Wdt, BL, 1024, 32, EpDt{dt_proj_b, dtb});

    // --- 6) selective scan (fused +xm*D and *silu(z)) ---
    scan_kernel<<<dim3(1024 / 256, Bn), 256, 0, stream>>>(
        dtb, xm, BC, sz, A_log, Dvec, yb);

    // --- 7) out_proj GEMM (M=BL, N=512, K=1024) + GRAPH scatter + residual ---
    wmma_gemm128<<<dim3(512 / 128, BL / 128), 256, 0, stream>>>(
        yb, Wout, BL, 512, 1024, EpOut{x, out});
}
